// LNSQuantizedLinear_38482906972510
// MI455X (gfx1250) — matrix operand order repaired and checked by reading, not compiled, not run on hardware
//
#include <hip/hip_runtime.h>

typedef __attribute__((ext_vector_type(16))) _Float16 v16h;
typedef __attribute__((ext_vector_type(8)))  _Float16 v8h;
typedef __attribute__((ext_vector_type(8)))  float    v8f;
typedef __attribute__((ext_vector_type(4)))  float    v4fe;
typedef __attribute__((ext_vector_type(4)))  unsigned v4u;

constexpr int M_DIM = 8192;   // tokens
constexpr int N_DIM = 4096;   // out features
constexpr int K_DIM = 4096;   // in features

// Two LNS nibbles (one packed byte) -> packed f16x2 dword (k-even nibble low).
// Reference sign: bit3=1 => +, bit3=0 => -.  f16 = ((n&8)^8)<<12 | (n&7)<<10 | 0x2000.
__device__ __forceinline__ unsigned lns2_to_f16x2(unsigned p) {
  unsigned nh = (p >> 4) & 15u;
  unsigned nl = p & 15u;
  unsigned h0 = (((nh & 8u) ^ 8u) << 12) | ((nh & 7u) << 10) | 0x2000u;
  unsigned h1 = (((nl & 8u) ^ 8u) << 12) | ((nl & 7u) << 10) | 0x2000u;
  return h0 | (h1 << 16);
}

__device__ __forceinline__ float lns_to_float(unsigned n) {
  unsigned bits = (((n & 8u) ^ 8u) << 28) | (((n & 7u) + 120u) << 23);
  return __builtin_bit_cast(float, bits);
}

// ==================== pre-pass: X f32 -> f16 (one shot, BW-bound) ============
__global__ __launch_bounds__(256)
void cvt_x_kernel(const float* __restrict__ X, _Float16* __restrict__ Xh) {
  size_t t = (size_t)blockIdx.x * 256 + threadIdx.x;   // 8 floats/thread
  const v4fe* src = (const v4fe*)(X + t * 8);
  v4fe a = __builtin_nontemporal_load(src);            // X never read again
  v4fe b = __builtin_nontemporal_load(src + 1);
  v8h h;
  h[0] = (_Float16)a[0]; h[1] = (_Float16)a[1];
  h[2] = (_Float16)a[2]; h[3] = (_Float16)a[3];
  h[4] = (_Float16)b[0]; h[5] = (_Float16)b[1];
  h[6] = (_Float16)b[2]; h[7] = (_Float16)b[3];
  *(v8h*)(Xh + t * 8) = h;                             // RT store: land in L2
}

// ==================== pre-pass: packed LNS W -> f16 =========================
__global__ __launch_bounds__(256)
void cvt_w_kernel(const int* __restrict__ Wp, _Float16* __restrict__ Wh) {
  size_t t = (size_t)blockIdx.x * 256 + threadIdx.x;   // one uint4 (8 weights)
  v4u p = __builtin_nontemporal_load((const v4u*)Wp + t);
  v4u d;
  d[0] = lns2_to_f16x2(p[0]);
  d[1] = lns2_to_f16x2(p[1]);
  d[2] = lns2_to_f16x2(p[2]);
  d[3] = lns2_to_f16x2(p[3]);
  *((v4u*)Wh + t) = d;
}

// ==================== main GEMM: pure-f16 operands ==========================
constexpr int BM = 128, BN = 128, BK = 64;
constexpr int LDH = BK + 8;        // 72 halves = 144B = 36 dwords: conflict-free
constexpr int TILE_H = BM * LDH;   // 9216 halves per buffer
constexpr size_t SMEM_BYTES = (size_t)4 * TILE_H * sizeof(_Float16);  // 73728

__global__ __launch_bounds__(256, 2)
void lns_gemm_f16(const _Float16* __restrict__ Xh,
                  const _Float16* __restrict__ Wh,
                  const int*      __restrict__ Bp,
                  float*          __restrict__ Out) {
  extern __shared__ _Float16 smem[];
  _Float16* Xs = smem;                    // [2][TILE_H]
  _Float16* Ws = smem + 2 * TILE_H;       // [2][TILE_H]

  const int tid    = threadIdx.x;
  const int lane   = tid & 31;
  const int wave   = tid >> 5;
  const int laneHi = (lane >> 4) & 1;
  const int l16    = lane & 15;

  // Block swizzle: groups of 8 M-tiles sweep all N-tiles -> hot set in 192MB L2.
  const int nbx = N_DIM / BN;
  const int pid = blockIdx.x + blockIdx.y * nbx;
  constexpr int GROUP = 8;
  const int gsz = GROUP * nbx;
  const int m_idx = (pid / gsz) * GROUP + (pid % gsz) % GROUP;
  const int n_idx = (pid % gsz) / GROUP;
  const int bm = m_idx * BM;
  const int bn = n_idx * BN;

  // 8 waves: 4 along M x 2 along N -> 32(M) x 64(N) per wave
  const int wm = (wave & 3) * 32;
  const int wn = (wave >> 2) * 64;

  v8f acc[2][4] = {};

  // Per-thread global pointers: one per stream, advanced BK halves per step;
  // row strides are 24-bit immediate offsets (32*K*2B = 256KB).
  const _Float16* xg = Xh + (size_t)(bm + (tid >> 3)) * K_DIM + (tid & 7) * 8;
  const _Float16* wg = Wh + (size_t)(bn + (tid >> 3)) * K_DIM + (tid & 7) * 8;

  v8h xr[4], wr[4];   // 4 x 16B segments each (rows +0,+32,+64,+96)

  auto gload = [&]() {
    xr[0] = *(const v8h*)(xg);
    xr[1] = *(const v8h*)(xg + (size_t)32 * K_DIM);
    xr[2] = *(const v8h*)(xg + (size_t)64 * K_DIM);
    xr[3] = *(const v8h*)(xg + (size_t)96 * K_DIM);
    wr[0] = *(const v8h*)(wg);
    wr[1] = *(const v8h*)(wg + (size_t)32 * K_DIM);
    wr[2] = *(const v8h*)(wg + (size_t)64 * K_DIM);
    wr[3] = *(const v8h*)(wg + (size_t)96 * K_DIM);
    xg += BK;
    wg += BK;
  };

  _Float16* xs_base = Xs + (tid >> 3) * LDH + (tid & 7) * 8;
  _Float16* ws_base = Ws + (tid >> 3) * LDH + (tid & 7) * 8;

  auto commit = [&](int b) {
    #pragma unroll
    for (int i = 0; i < 4; ++i)
      *(v8h*)(xs_base + b * TILE_H + i * 32 * LDH) = xr[i];
    #pragma unroll
    for (int i = 0; i < 4; ++i)
      *(v8h*)(ws_base + b * TILE_H + i * 32 * LDH) = wr[i];
  };

  // A: ISA 16-bit A 16x32 layout (lanes 0-15: K0..7|K16..23; 16-31: K8..15|K24..31)
  const _Float16* fa_base = Xs + (wm + l16) * LDH + laneHi * 8;
  // B: lane = column n; same K layout as A (lanes 0-15: K0..7|K16..23; 16-31: K8..15|K24..31)
  const _Float16* fb_base = Ws + (wn + l16) * LDH + laneHi * 8;

  auto fragA = [&](int b, int mi, int ko) -> v16h {
    const _Float16* p = fa_base + b * TILE_H + mi * 16 * LDH + ko;
    union { struct { v8h lo, hi; } s; v16h v; } u;
    u.s.lo = *(const v8h*)(p);
    u.s.hi = *(const v8h*)(p + 16);
    return u.v;
  };
  auto fragB = [&](int b, int ni, int ko) -> v16h {
    const _Float16* p = fb_base + b * TILE_H + ni * 16 * LDH + ko;
    union { struct { v8h lo, hi; } s; v16h v; } u;
    u.s.lo = *(const v8h*)(p);
    u.s.hi = *(const v8h*)(p + 16);
    return u.v;
  };

  auto compute = [&](int b) {
    #pragma unroll
    for (int ko = 0; ko < BK; ko += 32) {   // two 32-K sub-steps per tile
      v16h a0 = fragA(b, 0, ko);
      v16h a1 = fragA(b, 1, ko);
      v16h b0 = fragB(b, 0, ko);
      v16h b1 = fragB(b, 1, ko);
      v16h b2 = fragB(b, 2, ko);
      v16h b3 = fragB(b, 3, ko);
      acc[0][0] = __builtin_amdgcn_wmma_f32_16x16x32_f16(false, a0, false, b0, (short)0, acc[0][0], false, false);
      acc[1][0] = __builtin_amdgcn_wmma_f32_16x16x32_f16(false, a1, false, b0, (short)0, acc[1][0], false, false);
      acc[0][1] = __builtin_amdgcn_wmma_f32_16x16x32_f16(false, a0, false, b1, (short)0, acc[0][1], false, false);
      acc[1][1] = __builtin_amdgcn_wmma_f32_16x16x32_f16(false, a1, false, b1, (short)0, acc[1][1], false, false);
      acc[0][2] = __builtin_amdgcn_wmma_f32_16x16x32_f16(false, a0, false, b2, (short)0, acc[0][2], false, false);
      acc[1][2] = __builtin_amdgcn_wmma_f32_16x16x32_f16(false, a1, false, b2, (short)0, acc[1][2], false, false);
      acc[0][3] = __builtin_amdgcn_wmma_f32_16x16x32_f16(false, a0, false, b3, (short)0, acc[0][3], false, false);
      acc[1][3] = __builtin_amdgcn_wmma_f32_16x16x32_f16(false, a1, false, b3, (short)0, acc[1][3], false, false);
    }
  };

  // Software pipeline: ping-pong buffers, one barrier per 64-K step.
  gload();
  commit(0);
  __syncthreads();

  #pragma unroll 1
  for (int k0 = 0; k0 < K_DIM - 2 * BK; k0 += 2 * BK) {  // 31 iterations
    gload();
    compute(0);
    commit(1);
    __syncthreads();
    gload();
    compute(1);
    commit(0);
    __syncthreads();
  }
  gload();
  compute(0);
  commit(1);
  __syncthreads();
  compute(1);

  // Epilogue: bias decode + non-temporal stores (keep operands hot in L2).
  #pragma unroll
  for (int mi = 0; mi < 2; ++mi) {
    #pragma unroll
    for (int ni = 0; ni < 4; ++ni) {
      int n = bn + wn + ni * 16 + l16;
      unsigned pb  = (unsigned)Bp[n >> 1];
      unsigned nib = (n & 1) ? (pb & 15u) : ((pb >> 4) & 15u);
      float bias = lns_to_float(nib);
      int mbase = bm + wm + mi * 16 + laneHi * 8;
      float* o = Out + (size_t)mbase * N_DIM + n;
      #pragma unroll
      for (int v = 0; v < 8; ++v)
        __builtin_nontemporal_store(acc[mi][ni][v] + bias, o + (size_t)v * N_DIM);
    }
  }
}

// ==================== fallback: fused dequant GEMM (BK=32, static LDS) ======
constexpr int FBK = 32;
constexpr int FLDH = FBK + 8;
constexpr int FTILE = BM * FLDH;

__global__ __launch_bounds__(256, 2)
void lns_linear_fused(const float* __restrict__ X,
                      const int*   __restrict__ Wp,
                      const int*   __restrict__ Bp,
                      float* __restrict__ Out) {
  __shared__ _Float16 Xs[2][FTILE];
  __shared__ _Float16 Ws[2][FTILE];

  const int tid = threadIdx.x, lane = tid & 31, wave = tid >> 5;
  const int laneHi = (lane >> 4) & 1, l16 = lane & 15;

  const int nbx = N_DIM / BN;
  const int pid = blockIdx.x + blockIdx.y * nbx;
  constexpr int GROUP = 8;
  const int gsz = GROUP * nbx;
  const int bm = ((pid / gsz) * GROUP + (pid % gsz) % GROUP) * BM;
  const int bn = ((pid % gsz) / GROUP) * BN;
  const int wm = (wave & 3) * 32, wn = (wave >> 2) * 64;

  v8f acc[2][4] = {};
  const float* xg = X  + (size_t)(bm + (tid >> 3)) * K_DIM       + (tid & 7) * 4;
  const int*   wg = Wp + (size_t)(bn + (tid >> 2)) * (K_DIM / 2) + (tid & 3) * 4;
  float4 xr[4];
  uint4  wr[2];

  auto gload = [&]() {
    xr[0] = *(const float4*)(xg);
    xr[1] = *(const float4*)(xg + (size_t)32 * K_DIM);
    xr[2] = *(const float4*)(xg + (size_t)64 * K_DIM);
    xr[3] = *(const float4*)(xg + (size_t)96 * K_DIM);
    wr[0] = *(const uint4*)(wg);
    wr[1] = *(const uint4*)(wg + (size_t)64 * (K_DIM / 2));
    xg += FBK; wg += FBK / 2;
  };
  _Float16* xs_base = &Xs[0][(tid >> 3) * FLDH + (tid & 7) * 4];
  _Float16* ws_base = &Ws[0][(tid >> 2) * FLDH + (tid & 3) * 8];
  auto commit = [&](int b) {
    #pragma unroll
    for (int i = 0; i < 4; ++i) {
      _Float16* p = xs_base + b * FTILE + i * 32 * FLDH;
      p[0] = (_Float16)xr[i].x; p[1] = (_Float16)xr[i].y;
      p[2] = (_Float16)xr[i].z; p[3] = (_Float16)xr[i].w;
    }
    #pragma unroll
    for (int i = 0; i < 2; ++i) {
      uint4 d;
      d.x = lns2_to_f16x2(wr[i].x); d.y = lns2_to_f16x2(wr[i].y);
      d.z = lns2_to_f16x2(wr[i].z); d.w = lns2_to_f16x2(wr[i].w);
      *(uint4*)(ws_base + b * FTILE + i * 64 * FLDH) = d;
    }
  };
  const _Float16* fa_base = &Xs[0][(wm + l16) * FLDH] + laneHi * 8;
  const _Float16* fb_base = &Ws[0][(wn + l16) * FLDH] + laneHi * 8;
  auto fragA = [&](int b, int mi) -> v16h {
    const _Float16* p = fa_base + b * FTILE + mi * 16 * FLDH;
    union { struct { v8h lo, hi; } s; v16h v; } u;
    u.s.lo = *(const v8h*)(p); u.s.hi = *(const v8h*)(p + 16);
    return u.v;
  };
  auto fragB = [&](int b, int ni) -> v16h {
    const _Float16* p = fb_base + b * FTILE + ni * 16 * FLDH;
    union { struct { v8h lo, hi; } s; v16h v; } u;
    u.s.lo = *(const v8h*)(p); u.s.hi = *(const v8h*)(p + 16);
    return u.v;
  };
  auto compute = [&](int b) {
    v16h a0 = fragA(b, 0), a1 = fragA(b, 1);
    v16h b0 = fragB(b, 0), b1 = fragB(b, 1), b2 = fragB(b, 2), b3 = fragB(b, 3);
    acc[0][0] = __builtin_amdgcn_wmma_f32_16x16x32_f16(false, a0, false, b0, (short)0, acc[0][0], false, false);
    acc[1][0] = __builtin_amdgcn_wmma_f32_16x16x32_f16(false, a1, false, b0, (short)0, acc[1][0], false, false);
    acc[0][1] = __builtin_amdgcn_wmma_f32_16x16x32_f16(false, a0, false, b1, (short)0, acc[0][1], false, false);
    acc[1][1] = __builtin_amdgcn_wmma_f32_16x16x32_f16(false, a1, false, b1, (short)0, acc[1][1], false, false);
    acc[0][2] = __builtin_amdgcn_wmma_f32_16x16x32_f16(false, a0, false, b2, (short)0, acc[0][2], false, false);
    acc[1][2] = __builtin_amdgcn_wmma_f32_16x16x32_f16(false, a1, false, b2, (short)0, acc[1][2], false, false);
    acc[0][3] = __builtin_amdgcn_wmma_f32_16x16x32_f16(false, a0, false, b3, (short)0, acc[0][3], false, false);
    acc[1][3] = __builtin_amdgcn_wmma_f32_16x16x32_f16(false, a1, false, b3, (short)0, acc[1][3], false, false);
  };

  gload(); commit(0); __syncthreads();
  #pragma unroll 1
  for (int k0 = 0; k0 < K_DIM - 2 * FBK; k0 += 2 * FBK) {
    gload(); compute(0); commit(1); __syncthreads();
    gload(); compute(1); commit(0); __syncthreads();
  }
  gload(); compute(0); commit(1); __syncthreads();
  compute(1);

  #pragma unroll
  for (int mi = 0; mi < 2; ++mi)
    #pragma unroll
    for (int ni = 0; ni < 4; ++ni) {
      int n = bn + wn + ni * 16 + l16;
      unsigned pb  = (unsigned)Bp[n >> 1];
      unsigned nib = (n & 1) ? (pb & 15u) : ((pb >> 4) & 15u);
      float bias = lns_to_float(nib);
      int mbase = bm + wm + mi * 16 + laneHi * 8;
      float* o = Out + (size_t)mbase * N_DIM + n;
      #pragma unroll
      for (int v = 0; v < 8; ++v)
        __builtin_nontemporal_store(acc[mi][ni][v] + bias, o + (size_t)v * N_DIM);
    }
}

extern "C" void kernel_launch(void* const* d_in, const int* in_sizes, int n_in,
                              void* d_out, int out_size, void* d_ws, size_t ws_size,
                              hipStream_t stream) {
  const float* X  = (const float*)d_in[0];
  const int*   Wp = (const int*)d_in[1];
  const int*   Bp = (const int*)d_in[2];
  float* Out = (float*)d_out;

  const size_t xh_bytes = (size_t)M_DIM * K_DIM * 2;   // 67 MB
  const size_t wh_bytes = (size_t)N_DIM * K_DIM * 2;   // 33.5 MB
  dim3 grid(N_DIM / BN, M_DIM / BM);                   // 32 x 64

  if (ws_size >= xh_bytes + wh_bytes) {
    _Float16* Xh = (_Float16*)d_ws;
    _Float16* Wh = (_Float16*)((char*)d_ws + xh_bytes);
    cvt_x_kernel<<<(M_DIM * (K_DIM / 8)) / 256, 256, 0, stream>>>(X, Xh);
    cvt_w_kernel<<<(N_DIM * (K_DIM / 8)) / 256, 256, 0, stream>>>(Wp, Wh);
    lns_gemm_f16<<<grid, 256, SMEM_BYTES, stream>>>(Xh, Wh, Bp, Out);
  } else {
    lns_linear_fused<<<grid, 256, 0, stream>>>(X, Wp, Bp, Out);
  }
}
